// AlphaZeroNetwork_22325240004683
// MI455X (gfx1250) — matrix operand order repaired and checked by reading, not compiled, not run on hardware
//
#include <hip/hip_runtime.h>
#include <math.h>

// ---------------------------------------------------------------------------
// MI455X (gfx1250), wave32. bf16 WMMA (v_wmma_f32_16x16x32_bf16) for all GEMMs,
// TDM (tensor_load_to_lds) for weight/activation staging into LDS.
// One workgroup per board; all activations live in LDS (271KB < 320KB/WGP).
// ---------------------------------------------------------------------------

typedef __attribute__((ext_vector_type(16))) __bf16 bf16x16;
typedef __attribute__((ext_vector_type(8)))  __bf16 bf16x8;
typedef __attribute__((ext_vector_type(8)))  float  f32x8;
typedef __attribute__((ext_vector_type(4)))  unsigned int u32x4;
typedef __attribute__((ext_vector_type(8)))  int i32x8;
typedef __attribute__((ext_vector_type(4)))  int i32x4;

union ABFrag { bf16x16 v; struct { bf16x8 lo; bf16x8 hi; } h; };

#define NPAD  240   // 225 nodes padded to 15 M-tiles of 16
#define SH    104   // stride (elems) of 96-wide bf16 buffers (16B-aligned rows)
#define XNS   200   // stride of 192-wide xn buffer
#define STX   40    // stride of 32-wide buffers (x^T staging, W_in^T)

// ---- LDS byte offsets (all 16B aligned) -----------------------------------
#define OFF_H     0        // h   bf16 [240][SH]   49920
#define OFF_Y1    49920    // y1  bf16 [240][SH]   49920 (also x^T staging)
#define OFF_XN    99840    // xn  bf16 [240][XNS]  96000 (also f32 staging/scratch)
// per-block image (one TDM transfer of 60672 B):
#define OFF_W1T   195840   // W1^T bf16 [96][XNS]  38400
#define OFF_W2T   234240   // W2^T bf16 [96][SH]   19968
#define OFF_B1    254208   // f32 [96]
#define OFF_B2    254592   // f32 [96]
#define OFF_LNS   254976   // f32 [192]
#define OFF_LNB   255744   // f32 [192]
// static image (one TDM transfer of 14848 B):
#define OFF_WINT  256512   // W_in^T bf16 [96][STX] 7680
#define OFF_WP1T  264192   // Wp1^T  bf16 [32][SH]  6656
#define OFF_BIN   270848   // f32 [96]
#define OFF_BP1   271232   // f32 [32]
#define LDS_BYTES 271360

#define BLKIMG_BYTES  60672   // W1T + W2T + b1,b2,lns,lnb
#define STATIMG_BYTES 14848   // WINT + WP1T + b_in + bp1

// ---- workspace byte offsets -----------------------------------------------
#define WS_P1G     0u                       // [2048][7200] bf16 = 29,491,200
#define WS_WP2T    29491200u                // [240][7200]  bf16 =  3,456,000
#define WS_BLKIMG  32947200u                // 6 * 60672
#define WS_STATIMG 33311232u                // 14848

// ---------------------------------------------------------------------------
// TDM: 1D copy of n8 * 8 bytes global -> LDS via one tensor descriptor.
// Issued once per wave (EXEC ignored by tensor ops); caller gates to wave 0
// and must s_wait_tensorcnt before the consuming barrier.
// This toolchain exposes the 6-arg builtin:
//   (u32x4 g0, i32x8 g1, i32x4 g2, i32x4 g3, i32x8 extra, i32 cpol)
#if __has_builtin(__builtin_amdgcn_tensor_load_to_lds)
#define HAVE_TDM 1
__device__ __forceinline__ void tdm_copy_1d(void* lds_dst, const void* gsrc, int n8) {
  unsigned long long va = (unsigned long long)(uintptr_t)gsrc;
  unsigned int ldsa = (unsigned int)(uintptr_t)lds_dst;
  u32x4 g0;
  g0[0] = 1u;                                           // count=1 (valid D#)
  g0[1] = ldsa;                                         // lds_addr
  g0[2] = (unsigned int)(va & 0xFFFFFFFFu);             // global_addr[31:0]
  g0[3] = ((unsigned int)(va >> 32) & 0x01FFFFFFu) | 0x80000000u; // [56:32]+type=2
  i32x8 g1;
  g1[0] = 0x00030000;                                   // data_size=8B, mask=0
  g1[1] = (int)(((unsigned)n8 & 0xFFFFu) << 16);        // tensor_dim0[15:0]
  g1[2] = (int)((((unsigned)n8 >> 16) & 0xFFFFu) | (1u << 16)); // dim0 hi, tensor_dim1=1
  g1[3] = (int)(((unsigned)n8 & 0xFFFFu) << 16);        // tile_dim0
  g1[4] = 1;                                            // tile_dim1=1, tile_dim2=0
  g1[5] = n8;                                           // tensor_dim0_stride lo
  g1[6] = 0;
  g1[7] = 0;
  i32x4 z4 = {0, 0, 0, 0};
  i32x8 z8 = {0, 0, 0, 0, 0, 0, 0, 0};
  __builtin_amdgcn_tensor_load_to_lds(g0, g1, z4, z4, z8, 0);
}
__device__ __forceinline__ void tdm_wait() {
  __builtin_amdgcn_s_wait_tensorcnt((short)0);
}
#else
#define HAVE_TDM 0
#endif

// Cooperative fallback copy (16B chunks), also used if TDM builtin missing.
__device__ __forceinline__ void coop_copy(void* dst, const void* src, int bytes,
                                          int t, int nthr) {
  uint4* d = (uint4*)dst;
  const uint4* s = (const uint4*)src;
  for (int i = t; i < (bytes >> 4); i += nthr) d[i] = s[i];
}

// --- WMMA fragment loaders (layouts per CDNA5 ISA 7.12.2) -------------------
__device__ __forceinline__ bf16x16 ldA(const __bf16* buf, int stride, int row0,
                                       int k0, int lane) {
  const __bf16* p = buf + (row0 + (lane & 15)) * stride + k0 + ((lane >> 4) << 3);
  ABFrag f;
  f.h.lo = *(const bf16x8*)(p);
  f.h.hi = *(const bf16x8*)(p + 16);
  return f.v;
}
__device__ __forceinline__ bf16x16 ldB(const __bf16* wt, int stride, int n0,
                                       int k0, int lane) {
  const __bf16* p = wt + (n0 + (lane & 15)) * stride + k0 + ((lane >> 4) << 3);
  ABFrag f;
  f.h.lo = *(const bf16x8*)(p);
  f.h.hi = *(const bf16x8*)(p + 16);
  return f.v;
}

// ---------------------------------------------------------------------------
struct K1Args {
  const float* x;
  const int* nidx; const float* nmask;
  const float* Wv1; const float* bv1;
  const float* Wv2; const float* bv2;
  const char* blkimg;   // 6 * BLKIMG_BYTES, LDS-image format
  const char* statimg;  // STATIMG_BYTES
  __bf16* p1g;          // [2048][7200] bf16
  float* vout;          // [2048]
};

__global__ __launch_bounds__(256) void k1_gnn(K1Args A) {
  extern __shared__ char smem[];
  __bf16* sh_h   = (__bf16*)(smem + OFF_H);
  __bf16* sh_y1  = (__bf16*)(smem + OFF_Y1);
  __bf16* sh_xt  = sh_y1;                      // [240][STX], dead after GEMM0
  __bf16* sh_xn  = (__bf16*)(smem + OFF_XN);
  float*  sh_xf  = (float*)(smem + OFF_XN);    // f32 alias (x staging / scratch)
  __bf16* sh_w1  = (__bf16*)(smem + OFF_W1T);
  __bf16* sh_w2  = (__bf16*)(smem + OFF_W2T);
  float*  sh_b1  = (float*)(smem + OFF_B1);
  float*  sh_b2  = (float*)(smem + OFF_B2);
  float*  sh_lns = (float*)(smem + OFF_LNS);
  float*  sh_lnb = (float*)(smem + OFF_LNB);
  __bf16* sh_wi  = (__bf16*)(smem + OFF_WINT);
  __bf16* sh_wp  = (__bf16*)(smem + OFF_WP1T);
  float*  sh_bin = (float*)(smem + OFF_BIN);
  float*  sh_bp1 = (float*)(smem + OFF_BP1);

  const int t    = threadIdx.x;
  const int wave = t >> 5;
  const int lane = t & 31;
  const int col  = lane & 15;
  const int hi   = lane >> 4;
  const int b    = blockIdx.x;

  // ---- Phase 0: TDM-stage x + static weight image; zero xn pad rows --------
#if HAVE_TDM
  if (wave == 0) {
    tdm_copy_1d(smem + OFF_XN, (const char*)A.x + (size_t)b * 28800, 3600);
    tdm_copy_1d(smem + OFF_WINT, A.statimg, STATIMG_BYTES / 8);
  }
#else
  coop_copy(smem + OFF_XN, (const char*)A.x + (size_t)b * 28800, 28800, t, 256);
  coop_copy(smem + OFF_WINT, A.statimg, STATIMG_BYTES, t, 256);
#endif
  for (int i = t; i < 15 * XNS; i += 256) sh_xn[225 * XNS + i] = (__bf16)0.f;
#if HAVE_TDM
  if (wave == 0) tdm_wait();
#endif
  __syncthreads();

  // ---- Phase 0b: transpose x -> x^T bf16 [240][STX] ------------------------
  if (t < NPAD) {
    #pragma unroll
    for (int c = 0; c < 32; c++)
      sh_xt[t * STX + c] = (t < 225) ? (__bf16)sh_xf[c * 225 + t] : (__bf16)0.f;
  }
  __syncthreads();

  // ---- Phase 1: GEMM0  h = relu(x^T @ W_in + b_in), 15x6 tiles, K=32 -------
  for (int tt = wave; tt < 90; tt += 8) {
    int mt = tt % 15, nt = tt / 15;
    float bias = sh_bin[nt * 16 + col];
    f32x8 acc;
    #pragma unroll
    for (int r = 0; r < 8; r++) acc[r] = bias;
    bf16x16 a  = ldA(sh_xt, STX, mt * 16, 0, lane);
    bf16x16 bb = ldB(sh_wi, STX, nt * 16, 0, lane);
    acc = __builtin_amdgcn_wmma_f32_16x16x32_bf16(false, a, false, bb, (short)0,
                                                  acc, false, false);
    int rb = mt * 16 + hi * 8, cc = nt * 16 + col;
    #pragma unroll
    for (int r = 0; r < 8; r++)
      sh_h[(rb + r) * SH + cc] = (__bf16)__builtin_fmaxf(acc[r], 0.f);
  }
  __syncthreads();

  // ---- Phase 2: six graph blocks -------------------------------------------
  for (int blk = 0; blk < 6; blk++) {
    // 2a: one TDM transfer for the whole per-block LDS image
#if HAVE_TDM
    if (wave == 0) {
      tdm_copy_1d(smem + OFF_W1T, A.blkimg + blk * BLKIMG_BYTES, BLKIMG_BYTES / 8);
      tdm_wait();
    }
#else
    coop_copy(smem + OFF_W1T, A.blkimg + blk * BLKIMG_BYTES, BLKIMG_BYTES, t, 256);
#endif
    __syncthreads();

    // 2b: neighbor mean + LayerNorm, one node per thread, vectorized b128
    if (t < 225) {
      const int nd = t;
      int   idxv[8];
      float mk[8];
      #pragma unroll
      for (int j = 0; j < 8; j++) {
        int ix = A.nidx[nd * 8 + j];
        idxv[j] = ix < 0 ? 0 : ix;
        mk[j]   = A.nmask[nd * 8 + j];
      }
      float deg = 0.f;
      #pragma unroll
      for (int j = 0; j < 8; j++) deg += mk[j];
      float inv = 1.f / fmaxf(deg, 1.f);

      float s1 = 0.f, s2 = 0.f;
      #pragma unroll
      for (int cc = 0; cc < 96; cc += 8) {
        float acc8[8] = {0.f, 0.f, 0.f, 0.f, 0.f, 0.f, 0.f, 0.f};
        #pragma unroll
        for (int j = 0; j < 8; j++) {
          bf16x8 nv = *(const bf16x8*)&sh_h[idxv[j] * SH + cc];
          #pragma unroll
          for (int e = 0; e < 8; e++) acc8[e] += mk[j] * (float)nv[e];
        }
        bf16x8 hv = *(const bf16x8*)&sh_h[nd * SH + cc];
        bf16x8 nmout;
        #pragma unroll
        for (int e = 0; e < 8; e++) {
          float nmv = acc8[e] * inv;
          nmout[e] = (__bf16)nmv;
          float hh = (float)hv[e];
          s1 += nmv + hh;
          s2 += nmv * nmv + hh * hh;
        }
        *(bf16x8*)&sh_xn[nd * XNS + 96 + cc] = nmout;
      }
      float mu  = s1 * (1.f / 192.f);
      float var = s2 * (1.f / 192.f) - mu * mu;
      float rs  = rsqrtf(var + 1e-6f);
      #pragma unroll
      for (int cc = 0; cc < 96; cc += 8) {
        bf16x8 hv = *(const bf16x8*)&sh_h[nd * SH + cc];
        bf16x8 nv = *(const bf16x8*)&sh_xn[nd * XNS + 96 + cc];
        bf16x8 o1, o2;
        #pragma unroll
        for (int e = 0; e < 8; e++) {
          int c = cc + e;
          o1[e] = (__bf16)(((float)hv[e] - mu) * rs * sh_lns[c] + sh_lnb[c]);
          o2[e] = (__bf16)(((float)nv[e] - mu) * rs * sh_lns[96 + c] + sh_lnb[96 + c]);
        }
        *(bf16x8*)&sh_xn[nd * XNS + cc]      = o1;
        *(bf16x8*)&sh_xn[nd * XNS + 96 + cc] = o2;
      }
    }
    __syncthreads();

    // 2c: GEMM1  y1 = relu(xn @ W1 + b1), K=192, dual accumulator chains
    for (int tt = wave; tt < 90; tt += 8) {
      int mt = tt % 15, nt = tt / 15;
      float bias = sh_b1[nt * 16 + col];
      f32x8 acc0, acc1;
      #pragma unroll
      for (int r = 0; r < 8; r++) { acc0[r] = bias; acc1[r] = 0.f; }
      #pragma unroll
      for (int kk = 0; kk < 192; kk += 64) {
        bf16x16 a0 = ldA(sh_xn, XNS, mt * 16, kk, lane);
        bf16x16 b0 = ldB(sh_w1, XNS, nt * 16, kk, lane);
        acc0 = __builtin_amdgcn_wmma_f32_16x16x32_bf16(false, a0, false, b0, (short)0,
                                                       acc0, false, false);
        bf16x16 a1 = ldA(sh_xn, XNS, mt * 16, kk + 32, lane);
        bf16x16 b1 = ldB(sh_w1, XNS, nt * 16, kk + 32, lane);
        acc1 = __builtin_amdgcn_wmma_f32_16x16x32_bf16(false, a1, false, b1, (short)0,
                                                       acc1, false, false);
      }
      int rb = mt * 16 + hi * 8, cc = nt * 16 + col;
      #pragma unroll
      for (int r = 0; r < 8; r++)
        sh_y1[(rb + r) * SH + cc] = (__bf16)__builtin_fmaxf(acc0[r] + acc1[r], 0.f);
    }
    __syncthreads();

    // 2d: GEMM2  h += y1 @ W2 + b2, K=96
    for (int tt = wave; tt < 90; tt += 8) {
      int mt = tt % 15, nt = tt / 15;
      float bias = sh_b2[nt * 16 + col];
      f32x8 acc;
      #pragma unroll
      for (int r = 0; r < 8; r++) acc[r] = bias;
      #pragma unroll
      for (int k0 = 0; k0 < 96; k0 += 32) {
        bf16x16 a  = ldA(sh_y1, SH, mt * 16, k0, lane);
        bf16x16 bb = ldB(sh_w2, SH, nt * 16, k0, lane);
        acc = __builtin_amdgcn_wmma_f32_16x16x32_bf16(false, a, false, bb, (short)0,
                                                      acc, false, false);
      }
      int rb = mt * 16 + hi * 8, cc = nt * 16 + col;
      #pragma unroll
      for (int r = 0; r < 8; r++) {
        float old = (float)sh_h[(rb + r) * SH + cc];
        sh_h[(rb + r) * SH + cc] = (__bf16)(old + acc[r]);
      }
    }
    __syncthreads();
  }

  // ---- Phase 3: policy p1 = relu(h @ Wp1 + bp1), 15x2 tiles, K=96 ----------
  for (int tt = wave; tt < 30; tt += 8) {
    int mt = tt % 15, nt = tt / 15;
    float bias = sh_bp1[nt * 16 + col];
    f32x8 acc;
    #pragma unroll
    for (int r = 0; r < 8; r++) acc[r] = bias;
    #pragma unroll
    for (int k0 = 0; k0 < 96; k0 += 32) {
      bf16x16 a  = ldA(sh_h, SH, mt * 16, k0, lane);
      bf16x16 bb = ldB(sh_wp, SH, nt * 16, k0, lane);
      acc = __builtin_amdgcn_wmma_f32_16x16x32_bf16(false, a, false, bb, (short)0,
                                                    acc, false, false);
    }
    int rb = mt * 16 + hi * 8, cc = nt * 16 + col;
    #pragma unroll
    for (int r = 0; r < 8; r++)
      sh_y1[(rb + r) * SH + cc] = (__bf16)__builtin_fmaxf(acc[r], 0.f);
  }
  __syncthreads();

  // ---- Phase 4: export p1 (coalesced) + value head -------------------------
  for (int i = t; i < 7200; i += 256) {
    int n = i >> 5, c = i & 31;
    A.p1g[(size_t)b * 7200 + i] = sh_y1[n * SH + c];
  }
  if (t < 96) {            // hmean over 225 real nodes
    float s = 0.f;
    for (int n = 0; n < 225; n++) s += (float)sh_h[n * SH + t];
    sh_xf[t] = s * (1.f / 225.f);
  }
  __syncthreads();
  if (t < 128) {           // v1 = relu(hmean @ Wv1 + bv1)
    float s = A.bv1[t];
    for (int c = 0; c < 96; c++) s += sh_xf[c] * A.Wv1[c * 128 + t];
    sh_xf[96 + t] = s > 0.f ? s : 0.f;
  }
  __syncthreads();
  if (t == 0) {
    float s = A.bv2[0];
    for (int j = 0; j < 128; j++) s += sh_xf[96 + j] * A.Wv2[j];
    A.vout[b] = tanhf(s);
  }
}

// ---------------------------------------------------------------------------
// K0w: build LDS-format weight images (transposed bf16 + f32 const pools)
struct K0WArgs {
  const float* W_in; const float* Wp1;
  const float* W1[6]; const float* W2[6];
  const float* b1[6]; const float* b2[6];
  const float* lns[6]; const float* lnb[6];
  const float* b_in; const float* bp1;
  char* blkimg; char* statimg;
};

__global__ __launch_bounds__(256) void k0_weights(K0WArgs W) {
  int tid = blockIdx.x * 256 + threadIdx.x;
  int np  = gridDim.x * 256;
  __bf16* winT = (__bf16*)W.statimg;               // [96][STX]
  __bf16* wp1T = (__bf16*)(W.statimg + 7680);      // [32][SH]
  float*  sbin = (float*)(W.statimg + 14336);      // [96]
  float*  sbp1 = (float*)(W.statimg + 14720);      // [32]
  for (int i = tid; i < 96 * STX; i += np) {
    int d = i / STX, c = i % STX;
    winT[i] = (__bf16)((c < 32) ? W.W_in[c * 96 + d] : 0.f);
  }
  for (int i = tid; i < 32 * SH; i += np) {
    int d = i / SH, k = i % SH;
    wp1T[i] = (__bf16)((k < 96) ? W.Wp1[k * 32 + d] : 0.f);
  }
  for (int i = tid; i < 96; i += np) sbin[i] = W.b_in[i];
  for (int i = tid; i < 32; i += np) sbp1[i] = W.bp1[i];
  for (int blk = 0; blk < 6; blk++) {
    char* img = W.blkimg + blk * BLKIMG_BYTES;
    __bf16* w1T = (__bf16*)img;                    // [96][XNS]
    __bf16* w2T = (__bf16*)(img + 38400);          // [96][SH]
    float*  f   = (float*)(img + 58368);           // b1,b2,lns,lnb
    for (int i = tid; i < 96 * XNS; i += np) {
      int d = i / XNS, k = i % XNS;
      w1T[i] = (__bf16)((k < 192) ? W.W1[blk][k * 96 + d] : 0.f);
    }
    for (int i = tid; i < 96 * SH; i += np) {
      int d = i / SH, k = i % SH;
      w2T[i] = (__bf16)((k < 96) ? W.W2[blk][k * 96 + d] : 0.f);
    }
    for (int i = tid; i < 96; i += np)  { f[i] = W.b1[blk][i]; f[96 + i] = W.b2[blk][i]; }
    for (int i = tid; i < 192; i += np) { f[192 + i] = W.lns[blk][i]; f[384 + i] = W.lnb[blk][i]; }
  }
}

// ---------------------------------------------------------------------------
// K0t: transpose Wp2 [7200][225] f32 -> Wp2^T [240][7200] bf16 (pad rows zero)
__global__ __launch_bounds__(256) void k0_transpose(const float* __restrict__ wp2,
                                                    __bf16* __restrict__ wp2t) {
  int i = blockIdx.x * 256 + threadIdx.x;
  if (i >= 240 * 7200) return;
  int n = i / 7200, k = i % 7200;
  wp2t[i] = (__bf16)((n < 225) ? wp2[(size_t)k * 225 + n] : 0.f);
}

// ---------------------------------------------------------------------------
// K2: policy out = p1 [2048][7200] @ Wp2 + bp2 -> [2048][225]
__global__ __launch_bounds__(128) void k2_policy(const __bf16* __restrict__ p1g,
                                                 const __bf16* __restrict__ wp2t,
                                                 const float* __restrict__ bp2,
                                                 float* __restrict__ pout) {
  const int wave = threadIdx.x >> 5;
  const int lane = threadIdx.x & 31;
  const int col  = lane & 15;
  const int hi   = lane >> 4;
  const int m0   = blockIdx.x * 64 + wave * 16;
  const int n0   = blockIdx.y * 16;

  float bias = (n0 + col < 225) ? bp2[n0 + col] : 0.f;
  f32x8 acc0, acc1;
  #pragma unroll
  for (int r = 0; r < 8; r++) { acc0[r] = bias; acc1[r] = 0.f; }

  const __bf16* pa0 = p1g  + (size_t)(m0 + col) * 7200 + (hi << 3);
  const __bf16* pb0 = wp2t + (size_t)(n0 + col) * 7200 + (hi << 3);
  #pragma unroll 4
  for (int k0 = 0; k0 < 7168; k0 += 64) {
    ABFrag fa, fb;
    fa.h.lo = *(const bf16x8*)(pa0 + k0);
    fa.h.hi = *(const bf16x8*)(pa0 + k0 + 16);
    fb.h.lo = *(const bf16x8*)(pb0 + k0);
    fb.h.hi = *(const bf16x8*)(pb0 + k0 + 16);
    acc0 = __builtin_amdgcn_wmma_f32_16x16x32_bf16(false, fa.v, false, fb.v, (short)0,
                                                   acc0, false, false);
    fa.h.lo = *(const bf16x8*)(pa0 + k0 + 32);
    fa.h.hi = *(const bf16x8*)(pa0 + k0 + 48);
    fb.h.lo = *(const bf16x8*)(pb0 + k0 + 32);
    fb.h.hi = *(const bf16x8*)(pb0 + k0 + 48);
    acc1 = __builtin_amdgcn_wmma_f32_16x16x32_bf16(false, fa.v, false, fb.v, (short)0,
                                                   acc1, false, false);
  }
  {  // tail K step (7168..7200)
    ABFrag fa, fb;
    fa.h.lo = *(const bf16x8*)(pa0 + 7168);
    fa.h.hi = *(const bf16x8*)(pa0 + 7168 + 16);
    fb.h.lo = *(const bf16x8*)(pb0 + 7168);
    fb.h.hi = *(const bf16x8*)(pb0 + 7168 + 16);
    acc0 = __builtin_amdgcn_wmma_f32_16x16x32_bf16(false, fa.v, false, fb.v, (short)0,
                                                   acc0, false, false);
  }
  if (n0 + col < 225) {
    int row0 = m0 + hi * 8;
    #pragma unroll
    for (int r = 0; r < 8; r++)
      pout[(size_t)(row0 + r) * 225 + n0 + col] = acc0[r] + acc1[r];
  }
}

// ---------------------------------------------------------------------------
extern "C" void kernel_launch(void* const* d_in, const int* in_sizes, int n_in,
                              void* d_out, int out_size, void* d_ws, size_t ws_size,
                              hipStream_t stream) {
  (void)n_in; (void)out_size; (void)ws_size;
  auto F = [&](int i) { return (const float*)d_in[i]; };

  K1Args a;
  K0WArgs w;
  const float *Wp2, *bp2;
  if (in_sizes[0] == 1800) {
    // jax pytree flattening (dict keys sorted)
    a.nidx  = (const int*)d_in[0];
    a.nmask = F(1);
    w.W_in  = F(2);  w.Wp1 = F(3);  Wp2 = F(4);
    a.Wv1   = F(5);  a.Wv2 = F(6);  w.b_in = F(7);
    for (int i = 0; i < 6; i++) {
      int base = 8 + 6 * i;
      w.W1[i]  = F(base + 0); w.W2[i]  = F(base + 1);
      w.b1[i]  = F(base + 2); w.b2[i]  = F(base + 3);
      w.lnb[i] = F(base + 4); w.lns[i] = F(base + 5);
    }
    w.bp1 = F(44); bp2 = F(45); a.bv1 = F(46); a.bv2 = F(47);
    a.x   = F(48);
  } else {
    // dict insertion order
    a.x = F(0); w.W_in = F(1); w.b_in = F(2);
    for (int i = 0; i < 6; i++) {
      int base = 3 + 6 * i;
      w.lns[i] = F(base + 0); w.lnb[i] = F(base + 1);
      w.W1[i]  = F(base + 2); w.b1[i]  = F(base + 3);
      w.W2[i]  = F(base + 4); w.b2[i]  = F(base + 5);
    }
    w.Wp1 = F(39); w.bp1 = F(40); Wp2 = F(41); bp2 = F(42);
    a.Wv1 = F(43); a.bv1 = F(44); a.Wv2 = F(45); a.bv2 = F(46);
    a.nidx = (const int*)d_in[47]; a.nmask = F(48);
  }

  char* ws = (char*)d_ws;
  __bf16* p1g  = (__bf16*)(ws + WS_P1G);
  __bf16* wp2t = (__bf16*)(ws + WS_WP2T);
  w.blkimg  = ws + WS_BLKIMG;
  w.statimg = ws + WS_STATIMG;
  a.blkimg  = w.blkimg;
  a.statimg = w.statimg;
  a.p1g = p1g;
  float* pout = (float*)d_out;
  a.vout = pout + (size_t)2048 * 225;

  k0_weights<<<256, 256, 0, stream>>>(w);
  k0_transpose<<<(240 * 7200 + 255) / 256, 256, 0, stream>>>(Wp2, wp2t);
  k1_gnn<<<2048, 256, LDS_BYTES, stream>>>(a);
  k2_policy<<<dim3(32, 15), 128, 0, stream>>>(p1g, wp2t, bp2, pout);
}
